// HMMLayer_77232101916950
// MI455X (gfx1250) — matrix operand order repaired and checked by reading, not compiled, not run on hardware
//
#include <hip/hip_runtime.h>
#include <hip/hip_bf16.h>

// HMM forward-backward, B=128, T=1024, K=256, gfx1250 (MI455X).
// bwd: beta scan -> d_out. fwd: alpha scan + fused posterior (one step behind),
// in-place on d_out. bf16 WMMA / fp32 accumulate; P / P^T pinned in VGPRs.
// Carried state uses a one-step-stale row normalizer (row scale cancels in the
// output normalizations) -> 2 barriers per timestep. Fast v_rcp_f32 for
// sigmoid and normalizers; fold reads contiguous (transposed partials).

typedef __attribute__((ext_vector_type(16))) __bf16 v16bf;
typedef __attribute__((ext_vector_type(8)))  float  v8f;

#define TT 1024
#define KK 256
#define MT 16     // batch tile = WMMA M

__device__ __forceinline__ float sigmoidf_(float v) {
    return __builtin_amdgcn_rcpf(1.0f + __expf(-v));
}

// Sum across the 16 lanes of a half-wave (butterfly; all lanes get the sum).
__device__ __forceinline__ float hsum16_(float v) {
    v += __shfl_xor(v, 1, 32);
    v += __shfl_xor(v, 2, 32);
    v += __shfl_xor(v, 4, 32);
    v += __shfl_xor(v, 8, 32);
    return v;
}

// Sum 16 contiguous floats (4x float4 + tree adds), fixed order.
__device__ __forceinline__ float fold16_(const float* p) {
    const float4* q = (const float4*)p;
    float4 q0 = q[0], q1 = q[1], q2 = q[2], q3 = q[3];
    return (((q0.x + q0.y) + (q0.z + q0.w)) + ((q1.x + q1.y) + (q1.z + q1.w)))
         + (((q2.x + q2.y) + (q2.z + q2.w)) + ((q3.x + q3.y) + (q3.z + q3.w)));
}

union AView { v16bf v; uint4 q[2]; };

#define LOAD8(dst, arr, base)                                   \
    {   const float4* rp_ = (const float4*)&(arr)[(base)];      \
        float4 a_ = rp_[0], b_ = rp_[1];                        \
        dst[0]=a_.x; dst[1]=a_.y; dst[2]=a_.z; dst[3]=a_.w;     \
        dst[4]=b_.x; dst[5]=b_.y; dst[6]=b_.z; dst[7]=b_.w; }

// ---------------------------------------------------------------------------
// Backward: beta_{T-1}=1; beta_t = norm((y_{t+1}*beta_{t+1}) @ P^T) -> out.
// ---------------------------------------------------------------------------
__global__ __launch_bounds__(512) void hmm_bwd_kernel(
    const float* __restrict__ x, const float* __restrict__ P,
    float* __restrict__ out)
{
    const int tid = threadIdx.x, lane = tid & 31, hi = lane >> 4, nl = lane & 15;
    const int w = tid >> 5, wn = w << 4;
    const int b0 = blockIdx.x * MT;

    // B operand = P^T slice: B[k][n] = P[n][k]; lane n=l&15, elem e -> K=hi*16+e.
    v16bf Breg[8];
#pragma unroll
    for (int kb = 0; kb < 8; ++kb) {
        const float* prow = P + (size_t)(wn + nl) * KK + kb * 32 + hi * 16;
#pragma unroll
        for (int e = 0; e < 16; ++e) Breg[kb][e] = (__bf16)prow[e];
    }

    __shared__ __align__(16) __bf16 abuf[2][MT * KK];  // ping-pong carried state
    __shared__ __align__(16) float  wpart[MT][16];     // [row][wave]
    __shared__ __align__(16) float  rinv[MT];

    // t = T-1: beta = 1, state = y_{T-1}
#pragma unroll
    for (int i = 0; i < 8; ++i) {
        int j = tid * 8 + i, m = j >> 8, n = j & 255;
        size_t g = ((size_t)(b0 + m) * TT + (TT - 1)) * KK + n;
        abuf[0][j] = (__bf16)sigmoidf_(x[g]);
        out[g]  = 1.0f;
    }
    if (tid < MT) rinv[tid] = 1.0f;
    __syncthreads();

    int p = 0;
    for (int t = TT - 2; t >= 0; --t) {
        // --- recurrence: A loads + chained WMMAs ---
        const __bf16* ab = abuf[p];
        AView av[8];
#pragma unroll
        for (int kb = 0; kb < 8; ++kb) {
            const __bf16* ap = ab + nl * KK + kb * 32 + hi * 16;
            av[kb].q[0] = *(const uint4*)(ap);
            av[kb].q[1] = *(const uint4*)(ap + 8);
        }
        v8f c = {};
#pragma unroll
        for (int kb = 0; kb < 8; ++kb)
            c = __builtin_amdgcn_wmma_f32_16x16x32_bf16(
                    false, av[kb].v, false, Breg[kb], (short)0, c, false, false);

        // --- next-state write with STALE normalizer + row-sum partials ---
        float ris[8];  LOAD8(ris, rinv, hi * 8);
        float s[8];
        __bf16* aw = abuf[p ^ 1];
#pragma unroll
        for (int r = 0; r < 8; ++r) {
            int m = r + 8 * hi, n = wn + nl;
            size_t g = ((size_t)(b0 + m) * TT + t) * KK + n;
            aw[m * KK + n] = (__bf16)(c[r] * sigmoidf_(x[g]) * ris[r]);
            s[r] = hsum16_(c[r]);
        }
        if (nl == 0) {
#pragma unroll
            for (int r = 0; r < 8; ++r) wpart[r + 8 * hi][w] = s[r];
        }
        __syncthreads();                                   // B1
        if (tid < MT) rinv[tid] = __builtin_amdgcn_rcpf(fold16_(wpart[tid]));
        __syncthreads();                                   // B2

        // --- exact beta output (off the carried chain) ---
        float ri[8];  LOAD8(ri, rinv, hi * 8);
#pragma unroll
        for (int r = 0; r < 8; ++r) {
            int m = r + 8 * hi, n = wn + nl;
            size_t g = ((size_t)(b0 + m) * TT + t) * KK + n;
            out[g] = c[r] * ri[r];
            if (nl == 0 && t > 0) __builtin_prefetch(x + (g - KK), 0, 1);
        }
        p ^= 1;
    }
}

// ---------------------------------------------------------------------------
// Forward + posterior (one step behind): alpha scan; posterior in-place.
// ---------------------------------------------------------------------------
__global__ __launch_bounds__(512) void hmm_fwd_kernel(
    const float* __restrict__ x, const float* __restrict__ P,
    const float* __restrict__ p0, float* __restrict__ out)
{
    const int tid = threadIdx.x, lane = tid & 31, hi = lane >> 4, nl = lane & 15;
    const int w = tid >> 5, wn = w << 4;
    const int b0 = blockIdx.x * MT;

    // B operand = P slice: B[k][n] = P[k][n].
    v16bf Breg[8];
#pragma unroll
    for (int kb = 0; kb < 8; ++kb) {
#pragma unroll
        for (int e = 0; e < 16; ++e)
            Breg[kb][e] = (__bf16)P[(size_t)(kb * 32 + hi * 16 + e) * KK + wn + nl];
    }
    const float p0n = p0[wn + nl];

    __shared__ __align__(16) __bf16 abuf[2][MT * KK];
    __shared__ __align__(16) float  wpart2[32][16];  // rows 0-15: alpha, 16-31: g
    __shared__ __align__(16) float  rinv2[32];       // [0..15]=1/rowsum, [16..31]=1/gsum

    if (tid < MT) rinv2[tid] = 1.0f;
    // Warm L2 for t=0 rows (x and beta).
    if (nl == 0) {
#pragma unroll
        for (int r = 0; r < 8; ++r) {
            int m = r + 8 * hi;
            size_t g = ((size_t)(b0 + m) * TT) * KK + wn;
            __builtin_prefetch(x + g, 0, 1);
            __builtin_prefetch(out + g, 0, 1);
        }
    }
    __syncthreads();

    float gg_prev[8];
    int p = 0;
    for (int t = 0; t < TT; ++t) {
        // --- recurrence ---
        float v[8];
        if (t == 0) {
#pragma unroll
            for (int r = 0; r < 8; ++r) {
                int m = r + 8 * hi, n = wn + nl;
                size_t g = ((size_t)(b0 + m) * TT) * KK + n;
                v[r] = p0n * sigmoidf_(x[g]);
            }
        } else {
            const __bf16* ab = abuf[p];
            AView av[8];
#pragma unroll
            for (int kb = 0; kb < 8; ++kb) {
                const __bf16* ap = ab + nl * KK + kb * 32 + hi * 16;
                av[kb].q[0] = *(const uint4*)(ap);
                av[kb].q[1] = *(const uint4*)(ap + 8);
            }
            v8f c = {};
#pragma unroll
            for (int kb = 0; kb < 8; ++kb)
                c = __builtin_amdgcn_wmma_f32_16x16x32_bf16(
                        false, av[kb].v, false, Breg[kb], (short)0, c, false, false);
#pragma unroll
            for (int r = 0; r < 8; ++r) {
                int m = r + 8 * hi, n = wn + nl;
                size_t g = ((size_t)(b0 + m) * TT + t) * KK + n;
                v[r] = c[r] * sigmoidf_(x[g]);
            }
        }

        // --- next-state write (stale normalizer) + alpha row-sum partials ---
        float ris[8];  LOAD8(ris, rinv2, hi * 8);
        float s[8];
        __bf16* aw = abuf[p ^ 1];
#pragma unroll
        for (int r = 0; r < 8; ++r) {
            int m = r + 8 * hi, n = wn + nl;
            aw[m * KK + n] = (__bf16)(v[r] * ris[r]);
            s[r] = hsum16_(v[r]);
        }
        if (nl == 0) {
#pragma unroll
            for (int r = 0; r < 8; ++r) wpart2[r + 8 * hi][w] = s[r];
        }
        __syncthreads();                                   // B1
        // lanes 0-15 fold alpha sums; lanes 16-31 fold posterior sums (t-1).
        if (tid < 32) rinv2[tid] = __builtin_amdgcn_rcpf(fold16_(wpart2[tid]));
        __syncthreads();                                   // B2

        // --- shadow: exact alpha, posterior_{t-1} store, g-partials for t ---
        float ri[8];   LOAD8(ri, rinv2, hi * 8);
        float gg[8];
#pragma unroll
        for (int r = 0; r < 8; ++r) {
            int m = r + 8 * hi, n = wn + nl;
            size_t g = ((size_t)(b0 + m) * TT + t) * KK + n;
            float a = v[r] * ri[r];                        // exact alpha_t
            gg[r] = a * out[g];                            // alpha_t * beta_t
        }
        if (t > 0) {
            float rig[8];  LOAD8(rig, rinv2, 16 + hi * 8);
#pragma unroll
            for (int r = 0; r < 8; ++r) {
                int m = r + 8 * hi, n = wn + nl;
                size_t gp = ((size_t)(b0 + m) * TT + (t - 1)) * KK + n;
                out[gp] = gg_prev[r] * rig[r];             // posterior_{t-1}
            }
        }
        float s2[8];
#pragma unroll
        for (int r = 0; r < 8; ++r) s2[r] = hsum16_(gg[r]);
        if (nl == 0) {
#pragma unroll
            for (int r = 0; r < 8; ++r) wpart2[16 + r + 8 * hi][w] = s2[r];
        }
#pragma unroll
        for (int r = 0; r < 8; ++r) gg_prev[r] = gg[r];
        if (nl == 0 && t < TT - 1) {
#pragma unroll
            for (int r = 0; r < 8; ++r) {
                int m = r + 8 * hi;
                size_t g = ((size_t)(b0 + m) * TT + t + 1) * KK + wn;
                __builtin_prefetch(x + g, 0, 1);
                __builtin_prefetch(out + g, 0, 1);
            }
        }
        p ^= 1;
    }

    // --- epilogue: flush posterior_{T-1} ---
    __syncthreads();
    if (tid < MT) rinv2[16 + tid] = __builtin_amdgcn_rcpf(fold16_(wpart2[16 + tid]));
    __syncthreads();
    {
        float rig[8];  LOAD8(rig, rinv2, 16 + hi * 8);
#pragma unroll
        for (int r = 0; r < 8; ++r) {
            int m = r + 8 * hi, n = wn + nl;
            size_t gp = ((size_t)(b0 + m) * TT + (TT - 1)) * KK + n;
            out[gp] = gg_prev[r] * rig[r];
        }
    }
}

extern "C" void kernel_launch(void* const* d_in, const int* in_sizes, int n_in,
                              void* d_out, int out_size, void* d_ws, size_t ws_size,
                              hipStream_t stream) {
    (void)in_sizes; (void)n_in; (void)d_ws; (void)ws_size; (void)out_size;
    const float* x  = (const float*)d_in[0];
    const float* P  = (const float*)d_in[1];
    const float* p0 = (const float*)d_in[2];
    float* out = (float*)d_out;

    // B=128 -> 8 batch tiles of 16 rows; 512 threads = 16 wave32 per block.
    hmm_bwd_kernel<<<dim3(8), dim3(512), 0, stream>>>(x, P, out);
    hmm_fwd_kernel<<<dim3(8), dim3(512), 0, stream>>>(x, P, p0, out);
}
